// C4HierarchicalExecutor_62380105007265
// MI455X (gfx1250) — matrix operand order repaired and checked by reading, not compiled, not run on hardware
//
#include <hip/hip_runtime.h>
#include <hip/hip_bf16.h>

// ---------------------------------------------------------------------------
// C4HierarchicalExecutor on MI455X (gfx1250, wave32, WMMA).
//
// Math: score(m, a) = (1/sqrt(20)) * sum_k enc(m)_k * enc(a)_k, enc bits = +-10.
// Max logit is analytically SMAX = 2000/sqrt(20) (exact-match row always
// exists), so softmax = exp(s - SMAX) / sumexp with a single pass.
// Write weights do not depend on evolving memory -> all 64 write softmax
// denominators are computed in parallel; the sequential scan reduces to a
// per-cell 64-step affine chain.
//
// Scores are integer-valued <= 2000 -> exact in f16, so
// v_wmma_f32_16x16x32_f16 computes them exactly (K padded 20 -> 32 w/ zeros).
// ---------------------------------------------------------------------------

typedef __attribute__((ext_vector_type(16))) _Float16 v16h;
typedef __attribute__((ext_vector_type(8)))  float    v8f;

#define M_SIZE   (1 << 20)
#define NBITS    20
#define NQ       64          // read queries = write queries = 64

__device__ __constant__ const float INV_SQRT_D = 0.22360679774997896f; // 1/sqrt(20)
__device__ __constant__ const float SMAX_LOGIT = 447.21359549995793f;  // 2000/sqrt(20)

// d_ws accumulator layout (floats):
//   [  0.. 63]  SE_read  (softmax denominators, read queries)
//   [ 64..127]  SE_write (softmax denominators, write queries)
//   [128..191]  WS_read  (sum exp * memory, read queries)
#define ACC_SE_R   0
#define ACC_SE_W   64
#define ACC_WS_R   128
#define ACC_N      192

// ---- WMMA fragment builders (16x16x32 f16, wave32; ISA 7.12.2 layouts) ----

// A (keys) 16x32: lane&15 -> row M; lane<16 holds K {0..7,16..23},
// lane>=16 holds K {8..15,24..31} in its 16 halves.
__device__ inline v16h make_key_frag(int mbase, int lane) {
  const int m     = mbase + (lane & 15);
  const int khalf = (lane & 16) ? 8 : 0;
  v16h a;
#pragma unroll
  for (int e = 0; e < 16; ++e) {
    const int k = khalf + ((e < 8) ? e : (e + 8));
    _Float16 v = (_Float16)0.0f;                       // K-padding 20..31
    if (k < NBITS)
      v = ((m >> k) & 1) ? (_Float16)10.0f : (_Float16)(-10.0f);
    a[e] = v;
  }
  return a;
}

// B (queries) 32x16: lane&15 -> column N; lanes 0-15 hold K {0..7,16..23},
// lanes 16-31 hold K {8..15,24..31} (same order as A).
__device__ inline v16h make_q_frag(const int* __restrict__ addrs,
                                   int tile_base, int lane) {
  const int addr  = addrs[tile_base + (lane & 15)];
  const int kbase = (lane & 16) ? 8 : 0;
  v16h b;
#pragma unroll
  for (int e = 0; e < 16; ++e) {
    const int k = kbase + ((e < 8) ? e : (e + 8));
    _Float16 v = (_Float16)0.0f;
    if (k < NBITS)
      v = ((addr >> k) & 1) ? (_Float16)10.0f : (_Float16)(-10.0f);
    b[e] = v;
  }
  return b;
}

__device__ inline v8f wmma_scores(v16h a, v16h b) {
  v8f c = {};
  return __builtin_amdgcn_wmma_f32_16x16x32_f16(
      /*neg_a=*/false, a, /*neg_b=*/false, b,
      /*c_mod=*/(short)0, c, /*reuse_a=*/false, /*reuse_b=*/false);
}

// ---------------------------------------------------------------------------
__global__ void c4h_init(float* __restrict__ acc) {
  if (threadIdx.x < ACC_N) acc[threadIdx.x] = 0.0f;
}

// Pass 1: sum-exp denominators for all 128 queries + weighted read sums.
__global__ void __launch_bounds__(256)
c4h_reduce(const float* __restrict__ mem,
           const int*   __restrict__ raddr,
           const int*   __restrict__ waddr,
           float*       __restrict__ acc) {
  __shared__ float s_acc[ACC_N];
  const int tid = threadIdx.x;
  for (int i = tid; i < ACC_N; i += blockDim.x) s_acc[i] = 0.0f;
  __syncthreads();

  const int lane   = tid & 31;
  const int wave   = tid >> 5;
  const int gw     = blockIdx.x * (blockDim.x >> 5) + wave;
  const int nwaves = gridDim.x * (blockDim.x >> 5);

  // 8 query fragments: tiles 0..3 = read b-tiles, 4..7 = write j-tiles.
  v16h bq[8];
#pragma unroll
  for (int t = 0; t < 4; ++t) bq[t]     = make_q_frag(raddr, t * 16, lane);
#pragma unroll
  for (int t = 0; t < 4; ++t) bq[4 + t] = make_q_frag(waddr, t * 16, lane);

  float se[8] = {0.f, 0.f, 0.f, 0.f, 0.f, 0.f, 0.f, 0.f};
  float ws[4] = {0.f, 0.f, 0.f, 0.f};

  const int ntiles = M_SIZE / 16;
  for (int tile = gw; tile < ntiles; tile += nwaves) {
    const int  mbase = tile * 16;
    const v16h a     = make_key_frag(mbase, lane);

    // memory values for this lane's 8 D rows (contiguous, 32B aligned)
    const int    mrow = mbase + ((lane & 16) ? 8 : 0);
    const float4 mv0  = ((const float4*)(mem + mrow))[0];
    const float4 mv1  = ((const float4*)(mem + mrow))[1];
    const float  mv[8] = {mv0.x, mv0.y, mv0.z, mv0.w,
                          mv1.x, mv1.y, mv1.z, mv1.w};

#pragma unroll
    for (int ct = 0; ct < 8; ++ct) {
      const v8f d = wmma_scores(a, bq[ct]);
#pragma unroll
      for (int r = 0; r < 8; ++r) {
        const float e = __expf(d[r] * INV_SQRT_D - SMAX_LOGIT);
        se[ct] += e;
        if (ct < 4) ws[ct] += e * mv[r];
      }
    }
  }

  // lanes l and l^16 hold the same column (rows 0-7 vs 8-15): combine.
#pragma unroll
  for (int ct = 0; ct < 8; ++ct) se[ct] += __shfl_xor(se[ct], 16, 32);
#pragma unroll
  for (int ct = 0; ct < 4; ++ct) ws[ct] += __shfl_xor(ws[ct], 16, 32);

  if (lane < 16) {
#pragma unroll
    for (int ct = 0; ct < 4; ++ct) {
      atomicAdd(&s_acc[ACC_SE_R + ct * 16 + lane], se[ct]);
      atomicAdd(&s_acc[ACC_SE_W + ct * 16 + lane], se[4 + ct]);
      atomicAdd(&s_acc[ACC_WS_R + ct * 16 + lane], ws[ct]);
    }
  }
  __syncthreads();
  for (int i = tid; i < ACC_N; i += blockDim.x)
    atomicAdd(&acc[i], s_acc[i]);
}

// Pass 2: recompute write weights per 128-row chunk, transpose via padded
// LDS, run the 64-step affine chain per memory cell; block 0 emits reads.
__global__ void __launch_bounds__(256)
c4h_apply(const float* __restrict__ mem,
          const int*   __restrict__ waddr,
          const float* __restrict__ wval,
          const float* __restrict__ wen,
          const float* __restrict__ acc,
          float*       __restrict__ out) {
  __shared__ float w_lds[128 * 65];   // [m_local][j], stride 65: conflict-free
  __shared__ float v_lds[NQ];

  const int tid  = threadIdx.x;
  const int lane = tid & 31;
  const int wave = tid >> 5;

  if (tid < NQ) v_lds[tid] = wval[tid];

  // per-lane query fragments + combined scale enable[j]/SE_write[j]
  v16h  bw[4];
  float wf[4];
#pragma unroll
  for (int t = 0; t < 4; ++t) {
    bw[t] = make_q_frag(waddr, t * 16, lane);
    const int j = t * 16 + (lane & 15);
    wf[t] = wen[j] / acc[ACC_SE_W + j];
  }

  const int block_base = blockIdx.x * 128;
  const int local_base = wave * 16;               // 8 waves x 16 rows = 128
  const int mbase      = block_base + local_base;
  const v16h a         = make_key_frag(mbase, lane);
  const int  rbase     = local_base + ((lane & 16) ? 8 : 0);

#pragma unroll
  for (int ct = 0; ct < 4; ++ct) {
    const v8f d = wmma_scores(a, bw[ct]);
    const int j = ct * 16 + (lane & 15);
#pragma unroll
    for (int r = 0; r < 8; ++r) {
      const float w = __expf(d[r] * INV_SQRT_D - SMAX_LOGIT) * wf[ct];
      w_lds[(rbase + r) * 65 + j] = w;
    }
  }
  __syncthreads();

  if (tid < 128) {
    const int m = block_base + tid;
    float x = mem[m];
#pragma unroll 16
    for (int j = 0; j < NQ; ++j) {
      const float w = w_lds[tid * 65 + j];
      x = x * (1.0f - w) + v_lds[j] * w;   // matches reference expression
    }
    out[NQ + m] = x;
  }

  if (blockIdx.x == 0 && tid < NQ)
    out[tid] = acc[ACC_WS_R + tid] / acc[ACC_SE_R + tid];
}

// ---------------------------------------------------------------------------
extern "C" void kernel_launch(void* const* d_in, const int* in_sizes, int n_in,
                              void* d_out, int out_size, void* d_ws, size_t ws_size,
                              hipStream_t stream) {
  const float* memory = (const float*)d_in[0];   // [1M] f32
  const int*   raddr  = (const int*)  d_in[1];   // [64] int
  const int*   waddr  = (const int*)  d_in[2];   // [64] int
  const float* wval   = (const float*)d_in[3];   // [64] f32
  const float* wen    = (const float*)d_in[4];   // [64] f32
  float*       out    = (float*)d_out;           // [64 reads] ++ [1M memory]
  float*       acc    = (float*)d_ws;            // 192 floats of scratch

  (void)in_sizes; (void)n_in; (void)out_size; (void)ws_size;

  c4h_init<<<1, 256, 0, stream>>>(acc);
  c4h_reduce<<<512, 256, 0, stream>>>(memory, raddr, waddr, acc);
  c4h_apply<<<M_SIZE / 128, 256, 0, stream>>>(memory, waddr, wval, wen, acc, out);
}
